// _GATConvGlobalAtomAttention_56951266345489
// MI455X (gfx1250) — compile-verified, matrix operand order checked by reading, not run on hardware
//
#include <hip/hip_runtime.h>
#include <hip/hip_bf16.h>
#include <math.h>

#define N_ATOM   100000
#define N_BOND   400000
#define N_GLOBAL 5000
#define IN_DIM   128
#define OUT_DIM  128
#define HEADS    2
#define NHID     256            // OUT_DIM * HEADS
#define NEG_SLOPE 0.2f
#define BN_EPS   1e-5f

typedef __attribute__((ext_vector_type(16))) __bf16 v16bf;
typedef __attribute__((ext_vector_type(8)))  float  v8f;

__device__ __forceinline__ float lrelu(float x) { return x > 0.f ? x : NEG_SLOPE * x; }

__device__ __forceinline__ void atomicMaxF(float* addr, float v) {
    if (v >= 0.f) atomicMax((int*)addr, __float_as_int(v));
    else          atomicMin((unsigned int*)addr, __float_as_uint(v));
}

// ---------------- generic helpers ----------------
__global__ void k_zero(float* __restrict__ p, long long n) {
    long long i = (long long)blockIdx.x * blockDim.x + threadIdx.x;
    long long stride = (long long)gridDim.x * blockDim.x;
    for (; i < n; i += stride) p[i] = 0.f;
}

// ---------------- weight pack: f32 W[128,256] -> bf16 WMMA-B-fragment-major ----------
// Layout: frag f = (ks*16 + ctile); per lane a contiguous 16-bf16 (32B) chunk:
//   Wfrag[((f*32)+lane)*16 + j]
// matching the 16-bit B-matrix VGPR layout of v_wmma_f32_16x16x32_bf16:
//   n = lane&15, kb = (lane>>4)*8; j<8 -> K = ks*32+kb+j ; j>=8 -> K = ks*32+16+kb+(j-8)
__global__ void k_packW(const float* __restrict__ W, __bf16* __restrict__ Wfrag) {
    int i = blockIdx.x * blockDim.x + threadIdx.x;      // over 4*16*32*16 = 32768
    if (i >= IN_DIM * NHID) return;
    const int j     = i & 15;
    const int lane  = (i >> 4) & 31;
    const int ctile = (i >> 9) & 15;
    const int ks    = i >> 13;
    const int n  = lane & 15;
    const int kb = (lane >> 4) * 8;
    const int k  = (j < 8) ? (ks * 32 + kb + j) : (ks * 32 + 16 + kb + (j - 8));
    const int col = ctile * 16 + n;
    Wfrag[i] = (__bf16)W[k * NHID + col];
}

// ---------------- projection GEMM: X[M,128] @ W[128,256] -> Y[M,256] ----------------
// One block = one 16-row tile. 4 waves x 4 col-tiles of 16 = 256 cols.
__global__ __launch_bounds__(128) void k_proj(const float* __restrict__ X,
                                              const __bf16* __restrict__ Wfrag,
                                              float* __restrict__ Y, int M) {
    __shared__ __bf16 sA[16 * IN_DIM];
    const int rowbase = blockIdx.x * 16;
    const int t = threadIdx.x;

    // stage A tile (16x128) into LDS as bf16; zero-fill rows past M
    {
        const int r  = t >> 3;          // 0..15
        const int c0 = (t & 7) * 16;    // 16 elems each
        const int grow = rowbase + r;
        if (grow < M) {
            const float* src = X + (size_t)grow * IN_DIM + c0;
            #pragma unroll
            for (int j = 0; j < 16; ++j) sA[r * IN_DIM + c0 + j] = (__bf16)src[j];
        } else {
            #pragma unroll
            for (int j = 0; j < 16; ++j) sA[r * IN_DIM + c0 + j] = (__bf16)0.f;
        }
    }
    __syncthreads();

    const int wave = t >> 5;
    const int lane = t & 31;
    const int m    = lane & 15;
    const int kb   = (lane >> 4) * 8;   // lanes 16-31 hold K+8 / K+24 halves

    // hoist all 4 K-step A fragments from LDS (vectorizes to ds_load_b128)
    v16bf Af[4];
    #pragma unroll
    for (int ks = 0; ks < 4; ++ks) {
        #pragma unroll
        for (int j = 0; j < 8; ++j) {
            Af[ks][j]     = sA[m * IN_DIM + ks * 32 + kb + j];
            Af[ks][8 + j] = sA[m * IN_DIM + ks * 32 + 16 + kb + j];
        }
    }

    const v16bf* __restrict__ wf = (const v16bf*)Wfrag;

    #pragma unroll
    for (int ct = 0; ct < 4; ++ct) {
        const int ctile = wave * 4 + ct;
        v8f acc = {};
        #pragma unroll
        for (int ks = 0; ks < 4; ++ks) {
            // pre-packed fragment: one 32B contiguous load per lane
            const v16bf Bf = wf[(ks * 16 + ctile) * 32 + lane];
            acc = __builtin_amdgcn_wmma_f32_16x16x32_bf16(
                false, Af[ks], false, Bf, (short)0, acc, false, false);
        }
        const int half = lane >> 4;
        const int col = ctile * 16 + (lane & 15);
        #pragma unroll
        for (int v = 0; v < 8; ++v) {
            const int grow = rowbase + v + 8 * half;
            if (grow < M) Y[(size_t)grow * NHID + col] = acc[v];
        }
    }
}

// ---------------- attention dot products: out[row][head] = <Y_row_head, attn_head> ----
// one wave per row; lanes 0-15 -> head 0, lanes 16-31 -> head 1 (8 cols each lane)
__global__ void k_dots(const float* __restrict__ Y, int M,
                       const float* __restrict__ attnL, const float* __restrict__ attnR,
                       float* __restrict__ outL, float* __restrict__ outR) {
    const int wavesPerBlock = blockDim.x >> 5;
    const int row = blockIdx.x * wavesPerBlock + (threadIdx.x >> 5);
    if (row >= M) return;
    const int lane = threadIdx.x & 31;
    const int c0 = lane * 8;            // covers 256 cols across 32 lanes
    const float* yp = Y + (size_t)row * NHID + c0;
    float pl = 0.f, pr = 0.f;
    #pragma unroll
    for (int j = 0; j < 8; ++j) {
        const float v = yp[j];
        pl += v * attnL[c0 + j];
        if (attnR) pr += v * attnR[c0 + j];
    }
    // reduce within each 16-lane half
    #pragma unroll
    for (int s = 1; s < 16; s <<= 1) {
        pl += __shfl_xor(pl, s, 32);
        if (attnR) pr += __shfl_xor(pr, s, 32);
    }
    if ((lane & 15) == 0) {
        const int head = lane >> 4;
        outL[row * HEADS + head] = pl;
        if (attnR) outR[row * HEADS + head] = pr;
    }
}

// ---------------- per-atom global edge: eu_raw + emax init ----------------
__global__ void k_init_atom(const float* __restrict__ el_u, const float* __restrict__ er,
                            const int* __restrict__ g2a, float* __restrict__ eu,
                            float* __restrict__ emax) {
    int i = blockIdx.x * blockDim.x + threadIdx.x;      // over N_ATOM*HEADS
    if (i >= N_ATOM * HEADS) return;
    const int n = i >> 1, h = i & 1;
    const float v = lrelu(el_u[g2a[n] * HEADS + h] + er[i]);
    eu[i] = v;
    emax[i] = v;    // every atom has exactly one global edge -> valid init
}

// ---------------- edge pass 1: raw logits + segment max ----------------
__global__ void k_edge_max(const float* __restrict__ el_h, const float* __restrict__ el_e,
                           const float* __restrict__ er,
                           const int* __restrict__ src, const int* __restrict__ dst,
                           float* __restrict__ eh, float* __restrict__ ee,
                           float* __restrict__ emax) {
    int i = blockIdx.x * blockDim.x + threadIdx.x;      // over N_BOND*HEADS
    if (i >= N_BOND * HEADS) return;
    const int b = i >> 1, h = i & 1;
    const int s = src[b], d = dst[b];
    const float erd = er[d * HEADS + h];
    const float vh = lrelu(el_h[s * HEADS + h] + erd);
    const float ve = lrelu(el_e[i] + erd);
    eh[i] = vh;
    ee[i] = ve;
    atomicMaxF(&emax[d * HEADS + h], fmaxf(vh, ve));
}

// ---------------- per-atom exp + esum init ----------------
__global__ void k_atom_exp(float* __restrict__ eu, const float* __restrict__ emax,
                           float* __restrict__ esum) {
    int i = blockIdx.x * blockDim.x + threadIdx.x;
    if (i >= N_ATOM * HEADS) return;
    const float v = expf(eu[i] - emax[i]);
    eu[i] = v;
    esum[i] = v;        // runs before bond adds (stream order)
}

// ---------------- edge pass 2: exp + segment sum ----------------
__global__ void k_edge_exp(float* __restrict__ eh, float* __restrict__ ee,
                           const int* __restrict__ dst, const float* __restrict__ emax,
                           float* __restrict__ esum) {
    int i = blockIdx.x * blockDim.x + threadIdx.x;
    if (i >= N_BOND * HEADS) return;
    const int b = i >> 1, h = i & 1;
    const float m = emax[dst[b] * HEADS + h];
    const float vh = expf(eh[i] - m);
    const float ve = expf(ee[i] - m);
    eh[i] = vh;
    ee[i] = ve;
    atomicAdd(&esum[dst[b] * HEADS + h], vh + ve);
}

// ---------------- message scatter: one wave per bond ----------------
__global__ void k_scatter(const float* __restrict__ H, const float* __restrict__ E,
                          const float* __restrict__ eh, const float* __restrict__ ee,
                          const float* __restrict__ esum,
                          const int* __restrict__ src, const int* __restrict__ dst,
                          float* __restrict__ Out) {
    const int b = blockIdx.x * (blockDim.x >> 5) + (threadIdx.x >> 5);
    if (b >= N_BOND) return;
    const int lane = threadIdx.x & 31;
    const int head = lane >> 4;
    const int c = (lane & 15) * 8;              // 8 cols per lane within a head
    const int s = src[b], d = dst[b];
    const float inv = 1.f / esum[d * HEADS + head];
    const float ah = eh[b * HEADS + head] * inv;
    const float ae = ee[b * HEADS + head] * inv;
    const float* hp = H + (size_t)s * NHID + head * OUT_DIM + c;
    const float* ep = E + (size_t)b * NHID + head * OUT_DIM + c;
    float* op = Out + (size_t)d * NHID + head * OUT_DIM + c;
    #pragma unroll
    for (int j = 0; j < 8; ++j) atomicAdd(&op[j], hp[j] * ah + ep[j] * ae);
}

// ---------------- per-atom finalize: + global msg, head mean, BN stats ----------------
__global__ void k_atom_final(const float* __restrict__ Out, const float* __restrict__ U,
                             const float* __restrict__ eu, const float* __restrict__ esum,
                             const int* __restrict__ g2a,
                             float* __restrict__ hN,
                             float* __restrict__ colsum, float* __restrict__ colsq) {
    const int wavesPerBlock = blockDim.x >> 5;
    const int gw = blockIdx.x * wavesPerBlock + (threadIdx.x >> 5);
    const int nw = gridDim.x * wavesPerBlock;
    const int lane = threadIdx.x & 31;
    const int d0 = lane * 4;                    // 4 cols per lane, 128 total
    float sum[4] = {0.f, 0.f, 0.f, 0.f};
    float sq[4]  = {0.f, 0.f, 0.f, 0.f};
    for (int n = gw; n < N_ATOM; n += nw) {
        const int g = g2a[n];
        const float au0 = eu[n * HEADS + 0] / esum[n * HEADS + 0];
        const float au1 = eu[n * HEADS + 1] / esum[n * HEADS + 1];
        const float* orow = Out + (size_t)n * NHID;
        const float* urow = U + (size_t)g * NHID;
        #pragma unroll
        for (int j = 0; j < 4; ++j) {
            const int d = d0 + j;
            const float v0 = orow[d]           + urow[d]           * au0;
            const float v1 = orow[OUT_DIM + d] + urow[OUT_DIM + d] * au1;
            const float hv = 0.5f * (v0 + v1);
            hN[(size_t)n * OUT_DIM + d] = hv;
            sum[j] += hv;
            sq[j]  += hv * hv;
        }
    }
    #pragma unroll
    for (int j = 0; j < 4; ++j) {
        atomicAdd(&colsum[d0 + j], sum[j]);
        atomicAdd(&colsq[d0 + j],  sq[j]);
    }
}

// ---------------- batch-norm + relu ----------------
__global__ void k_bn(const float* __restrict__ hN,
                     const float* __restrict__ colsum, const float* __restrict__ colsq,
                     const float* __restrict__ gamma, const float* __restrict__ beta,
                     float* __restrict__ out) {
    long long i = (long long)blockIdx.x * blockDim.x + threadIdx.x;
    if (i >= (long long)N_ATOM * OUT_DIM) return;
    const int c = (int)(i & (OUT_DIM - 1));
    const float invN = 1.f / (float)N_ATOM;
    const float mean = colsum[c] * invN;
    const float var  = colsq[c] * invN - mean * mean;
    const float y = (hN[i] - mean) * rsqrtf(var + BN_EPS) * gamma[c] + beta[c];
    out[i] = fmaxf(y, 0.f);
}

extern "C" void kernel_launch(void* const* d_in, const int* in_sizes, int n_in,
                              void* d_out, int out_size, void* d_ws, size_t ws_size,
                              hipStream_t stream) {
    const float* atom_feats   = (const float*)d_in[0];
    const float* bond_feats   = (const float*)d_in[1];
    const float* global_feats = (const float*)d_in[2];
    const float* W_atom       = (const float*)d_in[3];
    const float* W_bond       = (const float*)d_in[4];
    const float* W_global     = (const float*)d_in[5];
    const float* attn_l       = (const float*)d_in[6];
    const float* attn_r       = (const float*)d_in[7];
    const float* bn_gamma     = (const float*)d_in[8];
    const float* bn_beta      = (const float*)d_in[9];
    const int*   bond_src     = (const int*)d_in[10];
    const int*   bond_dst     = (const int*)d_in[11];
    const int*   g2a_src      = (const int*)d_in[12];
    float* out = (float*)d_out;

    // workspace carve-up (256B aligned)
    char* w = (char*)d_ws;
    auto alloc = [&](size_t bytes) -> void* {
        void* p = (void*)w;
        w += (bytes + 255) & ~(size_t)255;
        return p;
    };
    __bf16* Wa_b = (__bf16*)alloc((size_t)IN_DIM * NHID * 2);
    __bf16* Wb_b = (__bf16*)alloc((size_t)IN_DIM * NHID * 2);
    __bf16* Wg_b = (__bf16*)alloc((size_t)IN_DIM * NHID * 2);
    float* H    = (float*)alloc((size_t)N_ATOM   * NHID * 4);
    float* E    = (float*)alloc((size_t)N_BOND   * NHID * 4);
    float* U    = (float*)alloc((size_t)N_GLOBAL * NHID * 4);
    float* el_h = (float*)alloc((size_t)N_ATOM   * HEADS * 4);
    float* er   = (float*)alloc((size_t)N_ATOM   * HEADS * 4);
    float* el_e = (float*)alloc((size_t)N_BOND   * HEADS * 4);
    float* el_u = (float*)alloc((size_t)N_GLOBAL * HEADS * 4);
    float* eu   = (float*)alloc((size_t)N_ATOM   * HEADS * 4);
    float* eh   = (float*)alloc((size_t)N_BOND   * HEADS * 4);
    float* ee   = (float*)alloc((size_t)N_BOND   * HEADS * 4);
    float* emax = (float*)alloc((size_t)N_ATOM   * HEADS * 4);
    float* esum = (float*)alloc((size_t)N_ATOM   * HEADS * 4);
    float* Out  = (float*)alloc((size_t)N_ATOM   * NHID * 4);
    float* hN   = (float*)alloc((size_t)N_ATOM   * OUT_DIM * 4);
    float* colsum = (float*)alloc(OUT_DIM * 4);
    float* colsq  = (float*)alloc(OUT_DIM * 4);

    // weights -> bf16, pre-swizzled into WMMA B-fragment order (L2-resident)
    const int nw_elems = IN_DIM * NHID;
    k_packW<<<(nw_elems + 255) / 256, 256, 0, stream>>>(W_atom,   Wa_b);
    k_packW<<<(nw_elems + 255) / 256, 256, 0, stream>>>(W_bond,   Wb_b);
    k_packW<<<(nw_elems + 255) / 256, 256, 0, stream>>>(W_global, Wg_b);

    // zero accumulators
    k_zero<<<2048, 256, 0, stream>>>(Out, (long long)N_ATOM * NHID);
    k_zero<<<1, 256, 0, stream>>>(colsum, OUT_DIM);
    k_zero<<<1, 256, 0, stream>>>(colsq, OUT_DIM);

    // WMMA projections
    k_proj<<<(N_ATOM   + 15) / 16, 128, 0, stream>>>(atom_feats,   Wa_b, H, N_ATOM);
    k_proj<<<(N_BOND   + 15) / 16, 128, 0, stream>>>(bond_feats,   Wb_b, E, N_BOND);
    k_proj<<<(N_GLOBAL + 15) / 16, 128, 0, stream>>>(global_feats, Wg_b, U, N_GLOBAL);

    // attention dots (8 waves/block = 8 rows/block)
    k_dots<<<(N_ATOM   + 7) / 8, 256, 0, stream>>>(H, N_ATOM,   attn_l, attn_r, el_h, er);
    k_dots<<<(N_BOND   + 7) / 8, 256, 0, stream>>>(E, N_BOND,   attn_l, nullptr, el_e, nullptr);
    k_dots<<<(N_GLOBAL + 7) / 8, 256, 0, stream>>>(U, N_GLOBAL, attn_l, nullptr, el_u, nullptr);

    // softmax over edges
    k_init_atom<<<(N_ATOM * HEADS + 255) / 256, 256, 0, stream>>>(el_u, er, g2a_src, eu, emax);
    k_edge_max<<<(N_BOND * HEADS + 255) / 256, 256, 0, stream>>>(
        el_h, el_e, er, bond_src, bond_dst, eh, ee, emax);
    k_atom_exp<<<(N_ATOM * HEADS + 255) / 256, 256, 0, stream>>>(eu, emax, esum);
    k_edge_exp<<<(N_BOND * HEADS + 255) / 256, 256, 0, stream>>>(eh, ee, bond_dst, emax, esum);

    // weighted message scatter (1 wave per bond)
    k_scatter<<<(N_BOND + 7) / 8, 256, 0, stream>>>(H, E, eh, ee, esum, bond_src, bond_dst, Out);

    // global message + head mean + BN stats
    k_atom_final<<<512, 256, 0, stream>>>(Out, U, eu, esum, g2a_src, hN, colsum, colsq);

    // batch-norm + relu -> output
    k_bn<<<(int)(((long long)N_ATOM * OUT_DIM + 255) / 256), 256, 0, stream>>>(
        hN, colsum, colsq, bn_gamma, bn_beta, out);
}
